// GCN_16114717295262
// MI455X (gfx1250) — matrix operand order repaired and checked by reading, not compiled, not run on hardware
//
#include <hip/hip_runtime.h>
#include <hip/hip_bf16.h>

#define BATCH 4
#define NNODE 4096
#define DMODEL 128

typedef __attribute__((ext_vector_type(16))) __bf16 v16bf;
typedef __attribute__((ext_vector_type(8)))  float  v8f;
typedef __attribute__((ext_vector_type(4)))  float  v4f;

__device__ __forceinline__ v16bf cvt16(v4f a, v4f b, v4f c, v4f d) {
  float t[16];
  *(v4f*)(t + 0) = a; *(v4f*)(t + 4) = b; *(v4f*)(t + 8) = c; *(v4f*)(t + 12) = d;
  v16bf r;
#pragma unroll
  for (int i = 0; i < 16; ++i) r[i] = (__bf16)t[i];
  return r;
}

// ---------------------------------------------------------------------------
// Kernel 1: d[b,n] = rsqrt(sum_j adj[b,n,j])   (268 MB streaming, RT hint so
// the *tail* of adj is left resident in the 192 MB L2 for kernel 3)
// ---------------------------------------------------------------------------
__global__ void k_rowsum(const float* __restrict__ adj, float* __restrict__ dscale) {
  const int row = blockIdx.x;                       // flat b*N + n
  const v4f* p = (const v4f*)(adj + (size_t)row * NNODE);
  float s = 0.f;
#pragma unroll
  for (int i = 0; i < 4; ++i) {
    v4f v = p[threadIdx.x + i * 256];
    s += v.x + v.y + v.z + v.w;
  }
#pragma unroll
  for (int off = 16; off > 0; off >>= 1) s += __shfl_xor(s, off, 32);
  __shared__ float red[8];
  if ((threadIdx.x & 31) == 0) red[threadIdx.x >> 5] = s;
  __syncthreads();
  if (threadIdx.x == 0) {
    float t = 0.f;
#pragma unroll
    for (int i = 0; i < 8; ++i) t += red[i];
    dscale[row] = (t > 0.f) ? rsqrtf(t) : 0.f;
  }
}

// ---------------------------------------------------------------------------
// Kernel 2: Yt[b][h][j] = dscale[b,j] * sum_d W[h][d] * x[b,j,d]   (bf16 out)
// d_j is constant per output column -> scale the accumulator in the epilogue.
// WMMA: A = W (M=h, K=d), B = x^T (K=d, N=j).  One WG = 128 h x 128 j.
// ---------------------------------------------------------------------------
__global__ void k_xw(const float* __restrict__ x, const float* __restrict__ W,
                     const float* __restrict__ dscale, __bf16* __restrict__ Yt) {
  const int b    = blockIdx.y;
  const int jt   = blockIdx.x;                      // 128-node tile
  const int wave = threadIdx.x >> 5;
  const int lane = threadIdx.x & 31;
  const int l16  = lane & 15;
  const int kg   = lane >> 4;

  const int   hrow  = wave * 16 + l16;              // A-fragment row (channel)
  const int   jbase = jt * 128;
  const float* wrow = W + (size_t)hrow * DMODEL;

  const float* xrow[8];
#pragma unroll
  for (int nf = 0; nf < 8; ++nf) {
    int j = jbase + nf * 16 + l16;                  // B-fragment column (node)
    xrow[nf] = x + ((size_t)b * NNODE + j) * DMODEL;
  }

  v8f acc[8] = {};

#pragma unroll
  for (int kc = 0; kc < DMODEL; kc += 32) {
    // A fragment (16x32 bf16): K[kg*16, kg*16+16) contiguous
    v16bf af = cvt16(*(const v4f*)(wrow + kc + kg * 16),
                     *(const v4f*)(wrow + kc + kg * 16 + 4),
                     *(const v4f*)(wrow + kc + kg * 16 + 8),
                     *(const v4f*)(wrow + kc + kg * 16 + 12));
#pragma unroll
    for (int nf = 0; nf < 8; ++nf) {
      // B fragment (32x16 bf16): lane=column, K[kg*16, kg*16+16) contiguous
      const float* bp = xrow[nf] + kc + kg * 16;
      v16bf bf = cvt16(*(const v4f*)(bp + 0),  *(const v4f*)(bp + 4),
                       *(const v4f*)(bp + 8),  *(const v4f*)(bp + 12));
      acc[nf] = __builtin_amdgcn_wmma_f32_16x16x32_bf16(
          false, af, false, bf, (short)0, acc[nf], false, false);
    }
  }

  // C layout: vgpr r -> M = r + kg*8, N = l16;  scale by d_j here.
#pragma unroll
  for (int nf = 0; nf < 8; ++nf) {
    int j = jbase + nf * 16 + l16;
    float dj = dscale[b * NNODE + j];
#pragma unroll
    for (int r = 0; r < 8; ++r) {
      int h = wave * 16 + r + kg * 8;
      Yt[((size_t)b * DMODEL + h) * NNODE + j] = (__bf16)(acc[nf][r] * dj);
    }
  }
}

// ---------------------------------------------------------------------------
// Kernel 3: out[b,i,h] = relu( d_i * sum_j adj[b,i,j] * Yt[b,h,j] + bias[h] )
// Tiles walked in REVERSE dispatch order so the adj tail left in L2 by
// k_rowsum is consumed first; adj loads are non-temporal (single use).
// A chunk is double-buffered in registers so the HBM loads for kc+32 stay in
// flight across the 8-wmma burst; B loads (L2-resident Yt) grouped in one clause.
// ---------------------------------------------------------------------------
__global__ void k_gcn(const float* __restrict__ adj, const __bf16* __restrict__ Yt,
                      const float* __restrict__ dscale, const float* __restrict__ bias,
                      float* __restrict__ out) {
  const int b    = (int)gridDim.y - 1 - (int)blockIdx.y;   // reverse order
  const int rt   = (int)gridDim.x - 1 - (int)blockIdx.x;   // reverse order
  const int wave = threadIdx.x >> 5;
  const int lane = threadIdx.x & 31;
  const int l16  = lane & 15;
  const int kg   = lane >> 4;

  const int rowbase = rt * 128 + wave * 16;
  const float* arow = adj + ((size_t)b * NNODE + rowbase + l16) * NNODE;

  const __bf16* bcol[8];
#pragma unroll
  for (int nf = 0; nf < 8; ++nf)
    bcol[nf] = Yt + ((size_t)b * DMODEL + nf * 16 + l16) * NNODE + kg * 16;

  v8f acc[8] = {};
  v4f cur0, cur1, cur2, cur3;

#define LOADA(d0, d1, d2, d3, kc)                                              \
  d0 = __builtin_nontemporal_load((const v4f*)(arow + (kc) + kg * 16));        \
  d1 = __builtin_nontemporal_load((const v4f*)(arow + (kc) + kg * 16 + 4));    \
  d2 = __builtin_nontemporal_load((const v4f*)(arow + (kc) + kg * 16 + 8));    \
  d3 = __builtin_nontemporal_load((const v4f*)(arow + (kc) + kg * 16 + 12));

  LOADA(cur0, cur1, cur2, cur3, 0)

#define GEMM_BODY(kc)                                                          \
  {                                                                            \
    v16bf bf[8];                                                               \
    _Pragma("unroll")                                                          \
    for (int nf = 0; nf < 8; ++nf) bf[nf] = *(const v16bf*)(bcol[nf] + (kc));  \
    v16bf af = cvt16(cur0, cur1, cur2, cur3);                                  \
    _Pragma("unroll")                                                          \
    for (int nf = 0; nf < 8; ++nf)                                             \
      acc[nf] = __builtin_amdgcn_wmma_f32_16x16x32_bf16(                       \
          false, af, false, bf[nf], (short)0, acc[nf], false, false);          \
  }

#pragma unroll 1
  for (int kc = 0; kc < NNODE - 32; kc += 32) {
    v4f nxt0, nxt1, nxt2, nxt3;
    LOADA(nxt0, nxt1, nxt2, nxt3, kc + 32)   // HBM loads stay in flight
    GEMM_BODY(kc)                            // waits only on L2-resident B
    cur0 = nxt0; cur1 = nxt1; cur2 = nxt2; cur3 = nxt3;
  }
  GEMM_BODY(NNODE - 32)                      // peeled last chunk

  // Epilogue: scale by d_i, add bias, relu, store fp32
#pragma unroll
  for (int r = 0; r < 8; ++r) {
    int row = rowbase + r + kg * 8;
    float dv = dscale[b * NNODE + row];
    float* orow = out + ((size_t)b * NNODE + row) * DMODEL;
#pragma unroll
    for (int nf = 0; nf < 8; ++nf) {
      int h = nf * 16 + l16;
      float v = acc[nf][r] * dv + bias[h];
      orow[h] = (v > 0.f) ? v : 0.f;
    }
  }
#undef LOADA
#undef GEMM_BODY
}

// ---------------------------------------------------------------------------
extern "C" void kernel_launch(void* const* d_in, const int* in_sizes, int n_in,
                              void* d_out, int out_size, void* d_ws, size_t ws_size,
                              hipStream_t stream) {
  const float* x    = (const float*)d_in[0];   // [B,N,128]
  const float* adj  = (const float*)d_in[1];   // [B,N,N]
  const float* W    = (const float*)d_in[2];   // [128,128]
  const float* bias = (const float*)d_in[3];   // [128]
  float* out = (float*)d_out;                  // [B,N,128]

  float*  dscale = (float*)d_ws;                                        // 64 KB
  __bf16* Yt     = (__bf16*)((char*)d_ws + (size_t)BATCH * NNODE * 4);  // 4 MB

  k_rowsum<<<BATCH * NNODE, 256, 0, stream>>>(adj, dscale);
  k_xw <<<dim3(NNODE / 128, BATCH), 256, 0, stream>>>(x, W, dscale, Yt);
  k_gcn<<<dim3(NNODE / 128, BATCH), 256, 0, stream>>>(adj, Yt, dscale, bias, out);
}
